// GATNet_13099650253555
// MI455X (gfx1250) — matrix operand order repaired and checked by reading, not compiled, not run on hardware
//
#include <hip/hip_runtime.h>
#include <hip/hip_bf16.h>

#define NN 50000
#define NEDGE 800000
#define ETOT 850000           // + self loops
#define LEAKY_ATT 0.2f
#define LEAKY_ACT 0.02f
#define BN_EPS 1e-5f

typedef __attribute__((ext_vector_type(16))) _Float16 v16h;
typedef __attribute__((ext_vector_type(8)))  float    v8f;

// ---------- monotone u32 key for float atomic-max ----------
__device__ __forceinline__ unsigned fkey(float x) {
  unsigned u = __float_as_uint(x);
  return (u & 0x80000000u) ? ~u : (u | 0x80000000u);
}
__device__ __forceinline__ float fdec(unsigned k) {
  unsigned u = (k & 0x80000000u) ? (k ^ 0x80000000u) : ~k;
  return __uint_as_float(u);
}

// ---------- pack W[K x Nc] f32 -> f16 WMMA B-fragment layout ----------
// Wp[((kt*ntn + nt)*32 + lane)*16 + j] = W[(kt*32 + 16*(lane/16) + j) * Nc + nt*16 + lane%16]
// (zero-padded for n >= Nc, so the GEMM hot loop has no guards)
__global__ void gat_pack_w(const float* __restrict__ W, _Float16* __restrict__ Wp,
                           int K, int Nc, int ntn) {
  int idx = blockIdx.x * blockDim.x + threadIdx.x;
  int total = (K >> 5) * ntn * 512;
  if (idx >= total) return;
  int kt   = idx / (ntn * 512);
  int rem  = idx - kt * (ntn * 512);
  int nt   = rem >> 9;
  int rem2 = rem & 511;
  int lane = rem2 >> 4, j = rem2 & 15;
  int g = lane >> 4;
  int n = nt * 16 + (lane & 15);
  int k = (kt << 5) + g * 16 + j;
  Wp[idx] = (_Float16)((n < Nc) ? W[(size_t)k * Nc + n] : 0.f);
}

// ---------- WMMA GEMM: H[M x Nc] = X[M x K] * W[K x Nc], f16 mul / f32 acc ----
// block = 32*ntn threads; grid.x = M/16. A tile staged in LDS once per block
// (shared by all waves), B read pre-packed: both fragments are single 32B vectors.
__global__ void gat_gemm_wmma(const float* __restrict__ X,
                              const _Float16* __restrict__ Wp,
                              float* __restrict__ H,
                              int K, int Nc, int ntn) {
  __shared__ __align__(32) _Float16 As[12 * 512];   // up to K=384 (12 k-tiles), 12 KB
  const int tid  = threadIdx.x;
  const int rowb = blockIdx.x * 16;
  const int nkt  = K >> 5;

  // cooperative f32->f16 convert of the 16-row A strip into fragment layout:
  // As[kt*512 + lane*16 + j] = X[(rowb + lane%16)*K + kt*32 + 16*(lane/16) + j]
  for (int idx = tid; idx < nkt * 512; idx += blockDim.x) {
    int kt   = idx >> 9;
    int rem  = idx & 511;
    int lane = rem >> 4, j = rem & 15;
    int g = lane >> 4, r = lane & 15;
    int k = (kt << 5) + g * 16 + j;
    As[idx] = (_Float16)X[(size_t)(rowb + r) * K + k];
  }
  __syncthreads();

  const int lane = tid & 31;
  const int wave = tid >> 5;
  const int g = lane >> 4, l16 = lane & 15;
  const int n = wave * 16 + l16;

  v8f acc = {};
  for (int kt = 0; kt < nkt; ++kt) {
    v16h a = *(const v16h*)&As[kt * 512 + lane * 16];
    v16h b = *(const v16h*)&Wp[((size_t)(kt * ntn + wave) * 32 + lane) * 16];
    acc = __builtin_amdgcn_wmma_f32_16x16x32_f16(false, a, false, b,
                                                 (short)0, acc, false, false);
  }
#pragma unroll
  for (int v = 0; v < 8; ++v) {
    int r = rowb + v + 8 * g;              // C/D layout: VGPR v, group g -> M = v + 8g
    if (n < Nc) H[(size_t)r * Nc + n] = acc[v];
  }
}

// ---------- per-node attention scores + per-node softmax state init ----------
__global__ void gat_attn_scores(const float* __restrict__ H, int Nc,
                                const float* __restrict__ a_src,
                                const float* __restrict__ a_dst,
                                float* __restrict__ as_, float* __restrict__ ad_,
                                unsigned* __restrict__ emax, float* __restrict__ denom) {
  int nidx = blockIdx.x * blockDim.x + threadIdx.x;
  if (nidx >= NN) return;
  const float* h = H + (size_t)nidx * Nc;
  float s = 0.f, d = 0.f;
  for (int c = 0; c < Nc; ++c) { float v = h[c]; s += v * a_src[c]; d += v * a_dst[c]; }
  as_[nidx] = s; ad_[nidx] = d;
  emax[nidx] = 0u;      // key(-inf) > 0, so 0 is a safe identity
  denom[nidx] = 0.f;
}

__global__ void gat_init_agg(unsigned* __restrict__ agg, int count) {
  int i = blockIdx.x * blockDim.x + threadIdx.x;
  if (i < count) agg[i] = 0u;
}

__device__ __forceinline__ void edge_sd(int e, const long long* __restrict__ ei,
                                        int& s, int& d) {
  if (e < NEDGE) { s = (int)ei[e]; d = (int)ei[(size_t)NEDGE + e]; }
  else           { s = d = e - NEDGE; }        // appended self loops
}

__global__ void gat_edge_max(const long long* __restrict__ ei,
                             const float* __restrict__ as_, const float* __restrict__ ad_,
                             unsigned* __restrict__ emax) {
  int e = blockIdx.x * blockDim.x + threadIdx.x;
  if (e >= ETOT) return;
  int s, d; edge_sd(e, ei, s, d);
  float ev = as_[s] + ad_[d];
  ev = ev > 0.f ? ev : LEAKY_ATT * ev;
  atomicMax(&emax[d], fkey(ev));
}

__global__ void gat_edge_expsum(const long long* __restrict__ ei,
                                const float* __restrict__ as_, const float* __restrict__ ad_,
                                const unsigned* __restrict__ emax,
                                float* __restrict__ denom, float* __restrict__ ee) {
  int e = blockIdx.x * blockDim.x + threadIdx.x;
  if (e >= ETOT) return;
  int s, d; edge_sd(e, ei, s, d);
  float ev = as_[s] + ad_[d];
  ev = ev > 0.f ? ev : LEAKY_ATT * ev;
  float x = __expf(ev - fdec(emax[d]));
  ee[e] = x;
  atomicAdd(&denom[d], x);
}

// alpha-weighted per-channel segment max (4 channels per thread)
__global__ void gat_edge_scatter(const long long* __restrict__ ei,
                                 const float* __restrict__ ee,
                                 const float* __restrict__ denom,
                                 const float* __restrict__ H, int Nc, int cpe,
                                 unsigned* __restrict__ agg) {
  int tid = blockIdx.x * blockDim.x + threadIdx.x;
  int e = tid / cpe;
  if (e >= ETOT) return;
  int cg = tid - e * cpe;
  int s, d; edge_sd(e, ei, s, d);
  float alpha = ee[e] / denom[d];
  const float* hs = H + (size_t)s * Nc;
  unsigned* ag = agg + (size_t)d * Nc;
#pragma unroll
  for (int j = 0; j < 4; ++j) {
    int c = cg * 4 + j;
    if (c < Nc) atomicMax(&ag[c], fkey(alpha * hs[c]));
  }
}

__global__ void gat_finalize(const unsigned* __restrict__ agg,
                             const float* __restrict__ bias,
                             float* __restrict__ dst, int ldd, int Nc, int count) {
  int i = blockIdx.x * blockDim.x + threadIdx.x;
  if (i >= count) return;
  int nidx = i / Nc, c = i - nidx * Nc;
  dst[(size_t)nidx * ldd + c] = fdec(agg[i]) + bias[c];
}

// ---------- batchnorm: block-per-channel stats, then elementwise ----------
__global__ void gat_col_stats(const float* __restrict__ X, int ld, int nrows,
                              float* __restrict__ mu, float* __restrict__ rsig) {
  int c = blockIdx.x;
  float s = 0.f, s2 = 0.f;
  for (int r = threadIdx.x; r < nrows; r += blockDim.x) {
    float v = X[(size_t)r * ld + c];
    s += v; s2 += v * v;
  }
  __shared__ float sh[256], sh2[256];
  sh[threadIdx.x] = s; sh2[threadIdx.x] = s2;
  __syncthreads();
  for (int off = blockDim.x >> 1; off > 0; off >>= 1) {
    if ((int)threadIdx.x < off) { sh[threadIdx.x] += sh[threadIdx.x + off];
                                  sh2[threadIdx.x] += sh2[threadIdx.x + off]; }
    __syncthreads();
  }
  if (threadIdx.x == 0) {
    float m = sh[0] / nrows;
    float var = sh2[0] / nrows - m * m;
    mu[c] = m;
    rsig[c] = rsqrtf(var + BN_EPS);
  }
}

__global__ void gat_bn_act(float* __restrict__ buf, int Nc,
                           const float* __restrict__ mu, const float* __restrict__ rsig,
                           const float* __restrict__ gamma, const float* __restrict__ beta,
                           int count) {
  int i = blockIdx.x * blockDim.x + threadIdx.x;
  if (i >= count) return;
  int c = i % Nc;
  float v = (buf[i] - mu[c]) * rsig[c] * gamma[c] + beta[c];
  buf[i] = v > 0.f ? v : LEAKY_ACT * v;
}

__global__ void gat_bcast(float* __restrict__ dst, int ldd,
                          const float* __restrict__ g, int Nc, int count) {
  int i = blockIdx.x * blockDim.x + threadIdx.x;
  if (i >= count) return;
  int nidx = i / Nc, c = i - nidx * Nc;
  dst[(size_t)nidx * ldd + c] = g[c];
}

__global__ void gat_tanh_half(const float* __restrict__ in, float* __restrict__ out, int count) {
  int i = blockIdx.x * blockDim.x + threadIdx.x;
  if (i < count) out[i] = tanhf(in[i]) * 0.5f;
}

// ---------- host-side one full GAT layer ----------
static void run_gat_layer(hipStream_t stream,
                          const float* xin, int K,
                          const float* W, const float* a_src, const float* a_dst,
                          const float* bias, int Nc,
                          float* H, _Float16* Wp, const long long* ei,
                          float* as_, float* ad_, unsigned* emax, float* denom,
                          float* ee, unsigned* agg,
                          float* dst, int ldd) {
  int ntn = (Nc + 15) / 16;
  int wtotal = (K >> 5) * ntn * 512;
  gat_pack_w<<<(wtotal + 255) / 256, 256, 0, stream>>>(W, Wp, K, Nc, ntn);
  gat_gemm_wmma<<<NN / 16, 32 * ntn, 0, stream>>>(xin, Wp, H, K, Nc, ntn);
  gat_attn_scores<<<(NN + 255) / 256, 256, 0, stream>>>(H, Nc, a_src, a_dst, as_, ad_, emax, denom);
  int aggcount = NN * Nc;
  gat_init_agg<<<(aggcount + 255) / 256, 256, 0, stream>>>(agg, aggcount);
  gat_edge_max<<<(ETOT + 255) / 256, 256, 0, stream>>>(ei, as_, ad_, emax);
  gat_edge_expsum<<<(ETOT + 255) / 256, 256, 0, stream>>>(ei, as_, ad_, emax, denom, ee);
  int cpe = (Nc + 3) / 4;
  long long tot = (long long)ETOT * cpe;
  gat_edge_scatter<<<(int)((tot + 255) / 256), 256, 0, stream>>>(ei, ee, denom, H, Nc, cpe, agg);
  gat_finalize<<<(aggcount + 255) / 256, 256, 0, stream>>>(agg, bias, dst, ldd, Nc, aggcount);
}

static void run_bn_act(hipStream_t stream, float* buf, int Nc,
                       const float* gamma, const float* beta, float* mu, float* rsig) {
  gat_col_stats<<<Nc, 256, 0, stream>>>(buf, Nc, NN, mu, rsig);
  gat_bn_act<<<(NN * Nc + 255) / 256, 256, 0, stream>>>(buf, Nc, mu, rsig, gamma, beta, NN * Nc);
}

extern "C" void kernel_launch(void* const* d_in, const int* in_sizes, int n_in,
                              void* d_out, int out_size, void* d_ws, size_t ws_size,
                              hipStream_t stream) {
  (void)in_sizes; (void)n_in; (void)out_size; (void)ws_size;

  const float*     x  = (const float*)d_in[0];
  const long long* ei = (const long long*)d_in[1];
  // pytree flatten, sorted dict keys per layer: +0 W, +1 a_dst, +2 a_src, +3 b, +4 beta, +5 gamma
  auto P = [&](int i) { return (const float*)d_in[i]; };
  const int HEAD = 2, SKIP = 32, GLOB = 38, TAIL = 50;

  // workspace carve (~184 MB)
  char* wsb = (char*)d_ws; size_t woff = 0;
  auto carve = [&](size_t bytes) -> void* {
    void* p = wsb + woff; woff += (bytes + 255) & ~(size_t)255; return p;
  };
  float*     hbuf   = (float*)carve((size_t)NN * 128 * 4);
  float*     io1    = (float*)carve((size_t)NN * 128 * 4);
  float*     io2    = (float*)carve((size_t)NN * 128 * 4);
  float*     concat = (float*)carve((size_t)NN * 384 * 4);
  unsigned*  agg    = (unsigned*)carve((size_t)NN * 128 * 4);
  float*     ee     = (float*)carve((size_t)ETOT * 4);
  float*     as_    = (float*)carve((size_t)NN * 4);
  float*     ad_    = (float*)carve((size_t)NN * 4);
  unsigned*  emax   = (unsigned*)carve((size_t)NN * 4);
  float*     denom  = (float*)carve((size_t)NN * 4);
  float*     mu     = (float*)carve(512 * 4);
  float*     rsig   = (float*)carve(512 * 4);
  float*     h3     = (float*)carve((size_t)NN * 4 * 4);
  _Float16*  Wp     = (_Float16*)carve(131072);          // packed weights (max 96 KB)

  // ---- head stack: 5 layers, BN+act on 0..3, last writes concat[:,0:128] ----
  const float* cur = x;
  for (int i = 0; i < 5; ++i) {
    int b0 = HEAD + 6 * i;
    float* dst = (i == 4) ? concat : io1;
    int ldd    = (i == 4) ? 384 : 128;
    run_gat_layer(stream, cur, 128, P(b0 + 0), P(b0 + 2), P(b0 + 1), P(b0 + 3), 128,
                  hbuf, Wp, ei, as_, ad_, emax, denom, ee, agg, dst, ldd);
    if (i < 4) { run_bn_act(stream, io1, 128, P(b0 + 5), P(b0 + 4), mu, rsig); cur = io1; }
  }

  // ---- skip stack: 1 layer, no BN, writes concat[:,128:256] ----
  {
    int b0 = SKIP;
    run_gat_layer(stream, x, 128, P(b0 + 0), P(b0 + 2), P(b0 + 1), P(b0 + 3), 128,
                  hbuf, Wp, ei, as_, ad_, emax, denom, ee, agg, concat + 128, 384);
  }

  // ---- global stack: 2 layers, then column mean -> broadcast concat[:,256:384] ----
  {
    int b0 = GLOB;
    run_gat_layer(stream, x, 128, P(b0 + 0), P(b0 + 2), P(b0 + 1), P(b0 + 3), 128,
                  hbuf, Wp, ei, as_, ad_, emax, denom, ee, agg, io1, 128);
    run_bn_act(stream, io1, 128, P(b0 + 5), P(b0 + 4), mu, rsig);
    int b1 = GLOB + 6;
    run_gat_layer(stream, io1, 128, P(b1 + 0), P(b1 + 2), P(b1 + 1), P(b1 + 3), 128,
                  hbuf, Wp, ei, as_, ad_, emax, denom, ee, agg, io2, 128);
    gat_col_stats<<<128, 256, 0, stream>>>(io2, 128, NN, mu, rsig);   // mu = column mean
    gat_bcast<<<(NN * 128 + 255) / 256, 256, 0, stream>>>(concat + 256, 384, mu, 128, NN * 128);
  }

  // ---- tail stack: (384->128 with BN+act) then (128->3), tanh * 0.5 ----
  {
    int b0 = TAIL;
    run_gat_layer(stream, concat, 384, P(b0 + 0), P(b0 + 2), P(b0 + 1), P(b0 + 3), 128,
                  hbuf, Wp, ei, as_, ad_, emax, denom, ee, agg, io1, 128);
    run_bn_act(stream, io1, 128, P(b0 + 5), P(b0 + 4), mu, rsig);
    int b1 = TAIL + 6;
    run_gat_layer(stream, io1, 128, P(b1 + 0), P(b1 + 2), P(b1 + 1), P(b1 + 3), 3,
                  h3, Wp, ei, as_, ad_, emax, denom, ee, agg, io2, 3);
    gat_tanh_half<<<(NN * 3 + 255) / 256, 256, 0, stream>>>(io2, (float*)d_out, NN * 3);
  }
}
